// VertexModule_13391708029604
// MI455X (gfx1250) — matrix operand order repaired and checked by reading, not compiled, not run on hardware
//
#include <hip/hip_runtime.h>
#include <hip/hip_bf16.h>

// ---------------------------------------------------------------------------
// Problem constants (from reference)
// ---------------------------------------------------------------------------
#define N_VERTEX 100000
#define N_EDGE   800000
#define HID      256          // d
#define K1       512          // 2d

typedef __attribute__((ext_vector_type(16))) __bf16 v16bf;
typedef __attribute__((ext_vector_type(8)))  __bf16 v8bf;
typedef __attribute__((ext_vector_type(8)))  float  v8f;

union FragAB { v16bf v; v8bf h[2]; };

// ---------------------------------------------------------------------------
// Kernel 0: zero the aggregation buffer (ws is poisoned; atomics accumulate)
// ---------------------------------------------------------------------------
__global__ void zero_f4(float4* __restrict__ p, long n4) {
    long i = (long)blockIdx.x * blockDim.x + threadIdx.x;
    if (i < n4) p[i] = make_float4(0.f, 0.f, 0.f, 0.f);
}

// ---------------------------------------------------------------------------
// Kernel 1: repack W[K,256] f32 row-major -> bf16 WMMA B-fragment-major.
// Fragment f = nt*ktiles + kt. Lane l of fragment holds column nt*16+(l&15),
// K octets at kt*32 + (l>>4)*8 and +16 more  (ISA B layout:
// lanes 0-15 = K 0..7,16..23, lanes 16-31 = K 8..15,24..31, 2 bf16 per VGPR).
// ---------------------------------------------------------------------------
__global__ void convert_w(const float* __restrict__ W, __bf16* __restrict__ dst,
                          int ktiles) {
    int gid = blockIdx.x * blockDim.x + threadIdx.x;
    int total = 16 * ktiles * 32;          // frags * lanes
    if (gid >= total) return;
    int lane = gid & 31;
    int f    = gid >> 5;
    int kt   = f % ktiles;
    int nt   = f / ktiles;
    int col  = nt * 16 + (lane & 15);
    int kbase = kt * 32 + (lane >> 4) * 8;
    __bf16* o = dst + (long)gid * 16;
#pragma unroll
    for (int e = 0; e < 16; ++e)
        o[e] = (__bf16)W[(long)(kbase + (e < 8 ? e : e + 8)) * HID + col];
}

// ---------------------------------------------------------------------------
// Kernel 2: scatter-sum edge features onto receiving vertices.
// aggr (102 MB) is L2-resident (192 MB L2) -> atomics resolve on-chip.
// One block per edge; thread t owns column t (coalesced load + atomic).
// ---------------------------------------------------------------------------
__global__ __launch_bounds__(256) void scatter_add(
        const float* __restrict__ ef, const long long* __restrict__ recv,
        float* __restrict__ aggr) {
    int e = blockIdx.x;
    int t = threadIdx.x;
    long long r = recv[e];
    float v = ef[(long)e * HID + t];
    __hip_atomic_fetch_add(aggr + (long)r * HID + t, v,
                           __ATOMIC_RELAXED, __HIP_MEMORY_SCOPE_AGENT);
}

// ---------------------------------------------------------------------------
// One GEMM stage of the fused MLP. A is 32 x KDIM bf16 in LDS (row-major,
// padded stride). Wave (mw, nq) computes rows mw*16..+15, n-tiles nq*4..+3.
// A fragment (ISA 16-bit A layout): lane l -> row l&15, K octets at
// kstep*32 + (l>>4)*8 and +16 more: two 16-byte LDS reads.
// B fragments come fragment-major from global (L2-resident weights).
// ---------------------------------------------------------------------------
template<int KDIM, int ASTRIDE, bool RELU, bool TOGLOBAL>
__device__ __forceinline__ void gemm_stage(
        const __bf16* __restrict__ Asrc, const __bf16* __restrict__ Wfrag,
        const float* __restrict__ bias, __bf16* __restrict__ Odst, int ostride,
        float* __restrict__ gout, long grow_base, int mw, int nq, int lane) {
    const int hi = lane >> 4;
    const int lm = lane & 15;
    constexpr int KT = KDIM / 32;

    v8f acc[4] = {};
#pragma unroll 4
    for (int k = 0; k < KT; ++k) {
        const __bf16* ap = Asrc + (mw * 16 + lm) * ASTRIDE + k * 32 + hi * 8;
        FragAB a;
        a.h[0] = *(const v8bf*)ap;
        a.h[1] = *(const v8bf*)(ap + 16);
#pragma unroll
        for (int t = 0; t < 4; ++t) {
            int nt = nq * 4 + t;
            const __bf16* bp = Wfrag + ((long)(nt * KT + k) * 32 + lane) * 16;
            FragAB b;
            b.h[0] = *(const v8bf*)bp;
            b.h[1] = *(const v8bf*)(bp + 8);
            acc[t] = __builtin_amdgcn_wmma_f32_16x16x32_bf16(
                false, a.v, false, b.v, (short)0, acc[t], false, false);
        }
    }

    // Epilogue: C layout -> VGPR v: lanes 0-15 M=v, lanes 16-31 M=v+8; N=lm.
#pragma unroll
    for (int t = 0; t < 4; ++t) {
        int col = (nq * 4 + t) * 16 + lm;
        float bv = bias[col];
#pragma unroll
        for (int v = 0; v < 8; ++v) {
            int m = mw * 16 + v + 8 * hi;
            float val = acc[t][v] + bv;
            if (RELU) val = fmaxf(val, 0.f);
            if (TOGLOBAL) gout[(grow_base + m) * HID + col] = val;
            else          Odst[m * ostride + col] = (__bf16)val;
        }
    }
}

// ---------------------------------------------------------------------------
// Kernel 3: fused 3-layer MLP. Block = 32 vertices, 8 waves.
// h never touches HBM: LDS ping-pong  Abuf -> Hbuf -> Abuf -> global.
// ---------------------------------------------------------------------------
__global__ __launch_bounds__(256) void fused_mlp(
        const float* __restrict__ vf, const float* __restrict__ aggr,
        const float* __restrict__ b1, const float* __restrict__ b2,
        const float* __restrict__ b3,
        const __bf16* __restrict__ W1f, const __bf16* __restrict__ W2f,
        const __bf16* __restrict__ W3f, float* __restrict__ out) {
    __shared__ __align__(16) __bf16 Abuf[32 * 520];   // 33,280 B  (K=512 rows)
    __shared__ __align__(16) __bf16 Hbuf[32 * 264];   // 16,896 B  (K=256 rows)

    const int tid  = threadIdx.x;
    const int wave = tid >> 5;
    const int lane = tid & 31;
    const int mw   = wave & 1;   // M sub-tile (rows mw*16 .. +15)
    const int nq   = wave >> 1;  // N quarter  (cols nq*64 .. +63)
    const long vbase = (long)blockIdx.x * 32;

    // Stage [X | aggr] (32 x 512 f32) -> bf16 LDS, float4-vectorized.
    for (int i = tid; i < 32 * 128; i += 256) {
        int row = i >> 7;
        int c   = (i & 127) * 4;
        const float* src = (c < HID)
            ? (vf   + (vbase + row) * HID + c)
            : (aggr + (vbase + row) * HID + (c - HID));
        float4 f = *(const float4*)src;
        __bf16* dst = Abuf + row * 520 + c;
        dst[0] = (__bf16)f.x; dst[1] = (__bf16)f.y;
        dst[2] = (__bf16)f.z; dst[3] = (__bf16)f.w;
    }
    __syncthreads();

    // L1: [32,512] @ W1 + b1, relu -> Hbuf
    gemm_stage<K1, 520, true, false>(Abuf, W1f, b1, Hbuf, 264,
                                     nullptr, 0, mw, nq, lane);
    __syncthreads();
    // L2: [32,256] @ W2 + b2, relu -> Abuf (cols 0..255)
    gemm_stage<HID, 264, true, false>(Hbuf, W2f, b2, Abuf, 520,
                                      nullptr, 0, mw, nq, lane);
    __syncthreads();
    // L3: [32,256] @ W3 + b3 -> global f32 output
    gemm_stage<HID, 520, false, true>(Abuf, W3f, b3, nullptr, 0,
                                      out, vbase, mw, nq, lane);
}

// ---------------------------------------------------------------------------
// Host-side launcher
// ---------------------------------------------------------------------------
extern "C" void kernel_launch(void* const* d_in, const int* in_sizes, int n_in,
                              void* d_out, int out_size, void* d_ws, size_t ws_size,
                              hipStream_t stream) {
    const float*      vf   = (const float*)d_in[0];
    const float*      ef   = (const float*)d_in[1];
    const long long*  eidx = (const long long*)d_in[2];   // int64 per reference
    const float*      W1   = (const float*)d_in[3];
    const float*      b1   = (const float*)d_in[4];
    const float*      W2   = (const float*)d_in[5];
    const float*      b2   = (const float*)d_in[6];
    const float*      W3   = (const float*)d_in[7];
    const float*      b3   = (const float*)d_in[8];
    float*            out  = (float*)d_out;

    // Workspace layout
    char* ws = (char*)d_ws;
    const long aggrBytes = (long)N_VERTEX * HID * sizeof(float);   // 102,400,000
    float*  aggr = (float*)ws;
    __bf16* W1f  = (__bf16*)(ws + aggrBytes);                      // 512*256 bf16
    __bf16* W2f  = (__bf16*)(ws + aggrBytes + (long)K1  * HID * 2);
    __bf16* W3f  = (__bf16*)(ws + aggrBytes + (long)K1  * HID * 2
                                            + (long)HID * HID * 2);

    // 1) zero aggr (float4-wide)
    const long n4 = (long)N_VERTEX * HID / 4;                      // 6.4M
    zero_f4<<<(int)((n4 + 255) / 256), 256, 0, stream>>>((float4*)aggr, n4);

    // 2) repack weights to bf16 WMMA fragments (L2-resident, 512 KB total)
    convert_w<<<32, 256, 0, stream>>>(W1, W1f, K1 / 32);   // 8192 threads
    convert_w<<<16, 256, 0, stream>>>(W2, W2f, HID / 32);  // 4096 threads
    convert_w<<<16, 256, 0, stream>>>(W3, W3f, HID / 32);

    // 3) edge scatter-sum (receivers = edge_index[1] = second E entries)
    scatter_add<<<N_EDGE, 256, 0, stream>>>(ef, eidx + N_EDGE, aggr);

    // 4) fused MLP, 32 vertices per block (100000 / 32 = 3125 exactly)
    fused_mlp<<<N_VERTEX / 32, 256, 0, stream>>>(vf, aggr, b1, b2, b3,
                                                 W1f, W2f, W3f, out);
}
